// LocationSensitiveAttention_1382979469321
// MI455X (gfx1250) — compile-verified, matrix operand order checked by reading, not run on hardware
//
#include <hip/hip_runtime.h>
#include <hip/hip_bf16.h>

// ---------------------------------------------------------------------------
// Location-sensitive attention for MI455X (gfx1250, wave32, WMMA).
// B=32, T=4096, DV=512, DQ=512, ATTN=128, FILTERS=32, KSIZE=31
// Fused flash-style context accumulation: values read from HBM exactly once.
// ---------------------------------------------------------------------------

#define BB   32
#define TT   4096
#define DVV  512
#define AA   128
#define FF   32
#define KS   31
#define NTILE (TT / 16)     // 256 tiles per batch row

typedef __attribute__((ext_vector_type(16))) __bf16 v16bf;
typedef __attribute__((ext_vector_type(8)))  float  v8f;

union Frag16 {          // 16 bf16 elements = 8 dwords = one WMMA A/B operand
  v16bf    v;
  unsigned u[8];
  uint4    q[2];
};

__device__ __forceinline__ unsigned short f2bf(float x) {
  unsigned u = __float_as_uint(x);
  u += 0x7FFFu + ((u >> 16) & 1u);      // round-to-nearest-even
  return (unsigned short)(u >> 16);
}
__device__ __forceinline__ unsigned packbf(float a, float b) {
  return (unsigned)f2bf(a) | ((unsigned)f2bf(b) << 16);
}

// ---------------------------------------------------------------------------
// Prepack wv_w [512,128] into WMMA bf16 B-operand layout.
// pk[(((ks*8 + nt)*32 + lane)*8 + j]:
//   lane holds column n = nt*16 + lane%16
//   lanes 0-15 hold K {0..7, 16..23}, lanes 16-31 hold K {8..15, 24..31}
//   dword j packs (K, K+1) with even K in the low half.
// ---------------------------------------------------------------------------
__global__ __launch_bounds__(256) void lsa_prepack_wv(
    const float* __restrict__ wv_w, unsigned* __restrict__ pk) {
  int tid = blockIdx.x * 256 + threadIdx.x;    // 0..32767
  int j   = tid & 7;
  int L   = (tid >> 3) & 31;
  int nt  = (tid >> 8) & 7;
  int ks  = tid >> 11;
  int n   = nt * 16 + (L & 15);
  int kb  = ks * 32 + ((L >> 4) ? 8 : 0) + ((j < 4) ? (2 * j) : (16 + 2 * (j - 4)));
  pk[tid] = packbf(wv_w[kb * AA + n], wv_w[(kb + 1) * AA + n]);
}

__global__ __launch_bounds__(256) void lsa_prepack_loc(
    const float* __restrict__ loc_w, unsigned* __restrict__ pk) {
  int tid = blockIdx.x * 256 + threadIdx.x;    // 0..2047
  int j   = tid & 7;
  int L   = (tid >> 3) & 31;
  int nt  = tid >> 8;
  int n   = nt * 16 + (L & 15);
  int kb  = ((L >> 4) ? 8 : 0) + ((j < 4) ? (2 * j) : (16 + 2 * (j - 4)));
  pk[tid] = packbf(loc_w[kb * AA + n], loc_w[(kb + 1) * AA + n]);
}

// ---------------------------------------------------------------------------
// qbias[b][a] = query[b]·wq_w[:,a] + wq_b + wv_b + loc_b + attention_bias
// ---------------------------------------------------------------------------
__global__ __launch_bounds__(128) void lsa_qbias(
    const float* __restrict__ query, const float* __restrict__ wq_w,
    const float* __restrict__ wq_b,  const float* __restrict__ wv_b,
    const float* __restrict__ loc_b, const float* __restrict__ attn_b,
    float* __restrict__ qbias) {
  int b = blockIdx.x, a = threadIdx.x;
  const float* q = query + b * 512;
  float s = 0.f;
  #pragma unroll 8
  for (int k = 0; k < 512; ++k) s += q[k] * wq_w[k * AA + a];
  qbias[b * AA + a] = s + wq_b[a] + wv_b[a] + loc_b[a] + attn_b[a];
}

// ---------------------------------------------------------------------------
// Score (+ fused local-softmax context partial) kernel.
// One workgroup = 16 timesteps of one batch row; 8 waves, wave w owns attn
// columns [16w, 16w+16).  scores[b][t] = sum_a tanh(pre)[t][a] * proj_w[a].
// If do_ctx: also emits m_tile, Z_tile = sum exp(s-m_tile), and
// ctx_tile[d] = sum_t exp(s_t-m_tile) * values[t][d]  (tile still hot in L2).
// ---------------------------------------------------------------------------
__global__ __launch_bounds__(256) void lsa_score(
    const float*    __restrict__ values,  const float* __restrict__ prev,
    const float*    __restrict__ conv_w,  const float* __restrict__ conv_b,
    const unsigned* __restrict__ packWV,  const unsigned* __restrict__ packLOC,
    const float*    __restrict__ qbias,   const float* __restrict__ proj_w,
    float*          __restrict__ scores,
    float*          __restrict__ mtile,   float* __restrict__ ztile,
    float*          __restrict__ ctxp,    int do_ctx) {
  __shared__ __align__(16) unsigned vals[16 * 260];  // 16x512 bf16, padded stride
  __shared__ __align__(16) unsigned fbf [16 * 20];   // 16x32 bf16 conv features
  __shared__ float pa[48];                           // alignment window (46 used)
  __shared__ float wpart[8][16];                     // per-wave row partials
  __shared__ float earr[16];                         // exp(s - m_tile)

  const int tid  = threadIdx.x;
  const int b    = blockIdx.x >> 8;    // 256 tiles per batch row
  const int tile = blockIdx.x & 255;
  const int t0   = tile * 16;

  // --- stage previous_alignment window (SAME padding, K=31 -> +/-15) ---
  if (tid < 46) {
    int idx = t0 - 15 + tid;
    pa[tid] = (idx >= 0 && idx < TT) ? prev[b * TT + idx] : 0.0f;
  }

  // --- stage values tile [16 x 512] f32 -> bf16 LDS (conflict-free stride) ---
  const float2* vsrc = (const float2*)(values + ((size_t)(b * TT + t0)) * DVV);
  #pragma unroll
  for (int it = 0; it < 16; ++it) {
    int idx = tid + it * 256;           // 0..4095 float2 elements
    int row = idx >> 8, c2 = idx & 255;
    float2 v = vsrc[row * 256 + c2];
    vals[row * 260 + c2] = packbf(v.x, v.y);
  }
  __syncthreads();

  // --- 31-tap conv over alignment: each thread -> 2 of the 16x32 features ---
  {
    int t  = tid >> 4;
    int c0 = (tid & 15) * 2;
    float s0 = conv_b[c0], s1 = conv_b[c0 + 1];
    #pragma unroll
    for (int j = 0; j < KS; ++j) {
      float p = pa[t + j];
      s0 += p * conv_w[j * FF + c0];
      s1 += p * conv_w[j * FF + c0 + 1];
    }
    fbf[t * 20 + (c0 >> 1)] = packbf(s0, s1);
  }
  __syncthreads();

  const int lane  = tid & 31;
  const int w     = tid >> 5;          // wave id == n-tile id
  const int rbase = lane & 15;         // A row / B column within tile
  const int hi    = lane >> 4;         // lane half selects K sub-block
  const int hi4   = hi * 4;            // dword offset of K half-chunk

  v8f acc = {};

  // --- proc_loc = f(16x32) @ loc_w(32x16 slice), one bf16 WMMA ---
  {
    Frag16 fa, lb;
    fa.q[0] = *(const uint4*)&fbf[rbase * 20 + hi4];
    fa.q[1] = *(const uint4*)&fbf[rbase * 20 + 8 + hi4];
    const uint4* bp = (const uint4*)(packLOC + ((size_t)(w * 32 + lane) * 8));
    lb.q[0] = bp[0];
    lb.q[1] = bp[1];
    acc = __builtin_amdgcn_wmma_f32_16x16x32_bf16(
        false, fa.v, false, lb.v, (short)0, acc, false, false);
  }

  // --- wv = values(16x512) @ wv_w(512x16 slice): 16 bf16 WMMAs, K=32 each ---
  for (int ks = 0; ks < 16; ++ks) {
    Frag16 av, bv;
    const unsigned* arow = &vals[rbase * 260 + ks * 16 + hi4];
    av.q[0] = *(const uint4*)(arow);
    av.q[1] = *(const uint4*)(arow + 8);
    const uint4* bp = (const uint4*)(packWV + ((size_t)((ks * 8 + w) * 32 + lane) * 8));
    bv.q[0] = bp[0];
    bv.q[1] = bp[1];
    if (ks + 1 < 16)   // keep the next B-panel warm in the near cache
      __builtin_prefetch(packWV + ((size_t)(((ks + 1) * 8 + w) * 32 + lane) * 8), 0, 3);
    acc = __builtin_amdgcn_wmma_f32_16x16x32_bf16(
        false, av.v, false, bv.v, (short)0, acc, false, false);
  }

  // --- epilogue: tanh, project, reduce over attn dim (deterministic) ---
  const int   n  = w * 16 + rbase;
  const float qb = qbias[b * AA + n];
  const float pw = proj_w[n];
  #pragma unroll
  for (int i = 0; i < 8; ++i) {
    // C layout: vgpr i -> row i (lanes 0-15) / row i+8 (lanes 16-31), col=lane%16
    float vv = tanhf(acc[i] + qb) * pw;
    vv += __shfl_xor(vv, 1, 16);
    vv += __shfl_xor(vv, 2, 16);
    vv += __shfl_xor(vv, 4, 16);
    vv += __shfl_xor(vv, 8, 16);
    if (rbase == 0) wpart[w][i + hi * 8] = vv;
  }
  __syncthreads();

  if (tid < 16) {
    float s = 0.f;
    #pragma unroll
    for (int ww = 0; ww < 8; ++ww) s += wpart[ww][tid];   // fixed order
    scores[b * TT + t0 + tid] = s;
    earr[tid] = s;            // reuse as staging for the serial pass below
  }
  __syncthreads();

  if (!do_ctx) return;

  // --- tile-local softmax stats (serial 16-elt pass: trivial, deterministic) ---
  if (tid == 0) {
    float m = earr[0];
    #pragma unroll
    for (int t = 1; t < 16; ++t) m = fmaxf(m, earr[t]);
    float z = 0.f;
    #pragma unroll
    for (int t = 0; t < 16; ++t) { float e = expf(earr[t] - m); earr[t] = e; z += e; }
    mtile[b * NTILE + tile] = m;
    ztile[b * NTILE + tile] = z;
  }
  __syncthreads();

  // --- ctx_tile[d] = sum_t e_t * values[t][d]; tile re-read hits L2 ---
  {
    float2 cacc; cacc.x = 0.f; cacc.y = 0.f;
    #pragma unroll
    for (int t = 0; t < 16; ++t) {
      float  e = earr[t];
      float2 x = vsrc[t * 256 + tid];
      cacc.x += e * x.x;
      cacc.y += e * x.y;
    }
    ((float2*)ctxp)[(size_t)(b * NTILE + tile) * 256 + tid] = cacc;
  }
}

// ---------------------------------------------------------------------------
// Fused path combine #1: per-b global softmax stats from tile stats, then
// write attention weights.  scale[tile] = exp(m_tile - m) cached for #2.
// ---------------------------------------------------------------------------
__global__ __launch_bounds__(256) void lsa_combine_stats(
    const float* __restrict__ scores, const float* __restrict__ mtile,
    const float* __restrict__ ztile,  float* __restrict__ scale,
    float* __restrict__ mz,           float* __restrict__ out_w) {
  __shared__ float red[256];
  const int b = blockIdx.x, tid = threadIdx.x;

  float m = mtile[b * NTILE + tid];               // one tile per thread
  red[tid] = m; __syncthreads();
  for (int off = 128; off > 0; off >>= 1) {
    if (tid < off) red[tid] = fmaxf(red[tid], red[tid + off]);
    __syncthreads();
  }
  const float mg = red[0]; __syncthreads();

  float sc = expf(m - mg);
  scale[b * NTILE + tid] = sc;
  red[tid] = ztile[b * NTILE + tid] * sc; __syncthreads();
  for (int off = 128; off > 0; off >>= 1) {
    if (tid < off) red[tid] += red[tid + off];
    __syncthreads();
  }
  const float inv = 1.0f / red[0];
  if (tid == 0) { mz[b * 2] = mg; mz[b * 2 + 1] = inv; }

  const float* s = scores + b * TT;
  for (int t = tid; t < TT; t += 256)
    out_w[b * TT + t] = expf(s[t] - mg) * inv;
}

// Fused path combine #2: ctx[b][d] = invZ * sum_tile scale[tile]*ctxp[tile][d]
__global__ __launch_bounds__(256) void lsa_combine_ctx(
    const float* __restrict__ ctxp, const float* __restrict__ scale,
    const float* __restrict__ mz,   float* __restrict__ out_ctx) {
  const int b = blockIdx.x, tid = threadIdx.x;
  const float invZ = mz[b * 2 + 1];
  const float2* p = (const float2*)ctxp;
  float2 s; s.x = 0.f; s.y = 0.f;
  for (int c = 0; c < NTILE; ++c) {              // fixed order
    float  sc = scale[b * NTILE + c];
    float2 x  = p[(size_t)(b * NTILE + c) * 256 + tid];
    s.x += sc * x.x; s.y += sc * x.y;
  }
  s.x *= invZ; s.y *= invZ;
  ((float2*)out_ctx)[b * 256 + tid] = s;
}

// ---------------------------------------------------------------------------
// Fallback path (small workspace): plain softmax + second values pass.
// ---------------------------------------------------------------------------
__global__ __launch_bounds__(256) void lsa_softmax(
    const float* __restrict__ scores, float* __restrict__ out_w) {
  __shared__ float red[256];
  const int b = blockIdx.x, tid = threadIdx.x;
  const float* s = scores + b * TT;

  float m = -3.402823466e+38f;
  for (int t = tid; t < TT; t += 256) m = fmaxf(m, s[t]);
  red[tid] = m; __syncthreads();
  for (int off = 128; off > 0; off >>= 1) {
    if (tid < off) red[tid] = fmaxf(red[tid], red[tid + off]);
    __syncthreads();
  }
  m = red[0]; __syncthreads();

  float sum = 0.f;
  for (int t = tid; t < TT; t += 256) sum += expf(s[t] - m);
  red[tid] = sum; __syncthreads();
  for (int off = 128; off > 0; off >>= 1) {
    if (tid < off) red[tid] += red[tid + off];
    __syncthreads();
  }
  const float inv = 1.0f / red[0];
  for (int t = tid; t < TT; t += 256)
    out_w[b * TT + t] = expf(s[t] - m) * inv;
}

__global__ __launch_bounds__(256) void lsa_ctx_partial(
    const float* __restrict__ values, const float* __restrict__ attn_w,
    float* __restrict__ partials) {
  const int b = blockIdx.x >> 5, chunk = blockIdx.x & 31, tid = threadIdx.x;
  const float2* v = (const float2*)(values + ((size_t)b * TT) * DVV);
  const int tbase = chunk * 128;
  float2 acc; acc.x = 0.f; acc.y = 0.f;
  for (int t = 0; t < 128; ++t) {
    float wt = attn_w[b * TT + tbase + t];
    float2 x = v[(size_t)(tbase + t) * 256 + tid];
    acc.x += wt * x.x;
    acc.y += wt * x.y;
  }
  ((float2*)partials)[(size_t)(b * 32 + chunk) * 256 + tid] = acc;
}

__global__ __launch_bounds__(256) void lsa_ctx_reduce(
    const float* __restrict__ partials, float* __restrict__ out_ctx) {
  const int b = blockIdx.x, tid = threadIdx.x;
  const float2* p = (const float2*)partials;
  float2 s; s.x = 0.f; s.y = 0.f;
  #pragma unroll
  for (int c = 0; c < 32; ++c) {
    float2 x = p[(size_t)(b * 32 + c) * 256 + tid];
    s.x += x.x; s.y += x.y;
  }
  ((float2*)out_ctx)[b * 256 + tid] = s;
}

// ---------------------------------------------------------------------------
extern "C" void kernel_launch(void* const* d_in, const int* in_sizes, int n_in,
                              void* d_out, int out_size, void* d_ws, size_t ws_size,
                              hipStream_t stream) {
  const float* query  = (const float*)d_in[0];
  const float* values = (const float*)d_in[1];
  const float* prev   = (const float*)d_in[2];
  const float* conv_w = (const float*)d_in[3];
  const float* conv_b = (const float*)d_in[4];
  const float* loc_w  = (const float*)d_in[5];
  const float* loc_b  = (const float*)d_in[6];
  const float* wq_w   = (const float*)d_in[7];
  const float* wq_b   = (const float*)d_in[8];
  const float* wv_w   = (const float*)d_in[9];
  const float* wv_b   = (const float*)d_in[10];
  const float* attn_b = (const float*)d_in[11];
  const float* proj_w = (const float*)d_in[12];

  float* out_ctx = (float*)d_out;               // [32, 512]
  float* out_w   = out_ctx + BB * DVV;          // [32, 4096]

  unsigned char* ws = (unsigned char*)d_ws;
  unsigned* packWV  = (unsigned*)(ws);            // 32768 u32  (128 KB)
  unsigned* packLOC = (unsigned*)(ws + 131072);   //  2048 u32  (  8 KB)
  float*    qbias   = (float*)(ws + 139264);      //  4096 f32  ( 16 KB)
  float*    scores  = (float*)(ws + 155648);      // 131072 f32 (512 KB)
  // fused-path extras
  float*    mtile   = (float*)(ws + 679936);      //  8192 f32 ( 32 KB)
  float*    ztile   = (float*)(ws + 712704);      //  8192 f32 ( 32 KB)
  float*    scale   = (float*)(ws + 745472);      //  8192 f32 ( 32 KB)
  float*    mz      = (float*)(ws + 778240);      //    64 f32
  float*    ctxp    = (float*)(ws + 778752);      // 32*256*512 f32 (16 MB)
  const size_t fused_need = 778752u + (size_t)BB * NTILE * DVV * 4u;
  // fallback-path extra (overlaps the fused region)
  float*    parts   = (float*)(ws + 679936);      // 524288 f32 (2 MB)
  const int fused = (ws_size >= fused_need) ? 1 : 0;

  lsa_prepack_wv <<<128, 256, 0, stream>>>(wv_w, packWV);
  lsa_prepack_loc<<<  8, 256, 0, stream>>>(loc_w, packLOC);
  lsa_qbias      <<< BB, 128, 0, stream>>>(query, wq_w, wq_b, wv_b, loc_b, attn_b, qbias);

  lsa_score<<<BB * NTILE, 256, 0, stream>>>(
      values, prev, conv_w, conv_b, packWV, packLOC, qbias, proj_w,
      scores, mtile, ztile, ctxp, fused);

  if (fused) {
    lsa_combine_stats<<<BB, 256, 0, stream>>>(scores, mtile, ztile, scale, mz, out_w);
    lsa_combine_ctx  <<<BB, 256, 0, stream>>>(ctxp, scale, mz, out_ctx);
  } else {
    lsa_softmax    <<< BB, 256, 0, stream>>>(scores, out_w);
    lsa_ctx_partial<<<BB * 32, 256, 0, stream>>>(values, out_w, parts);
    lsa_ctx_reduce <<< BB, 256, 0, stream>>>(parts, out_ctx);
  }
}
